// XCA_50328426774879
// MI455X (gfx1250) — matrix operand order repaired and checked by reading, not compiled, not run on hardware
//
#include <hip/hip_runtime.h>
#include <hip/hip_bf16.h>

typedef __bf16 bf16;
typedef __attribute__((ext_vector_type(8)))  __bf16 bf16x8;
typedef __attribute__((ext_vector_type(16))) __bf16 bf16x16;
typedef __attribute__((ext_vector_type(8)))  float  f32x8;
typedef __attribute__((ext_vector_type(4)))  int    i32x4;

#define AS_GLOBAL __attribute__((address_space(1)))
#define AS_LDS    __attribute__((address_space(3)))

// ---------------------------------------------------------------------------
// CDNA5 async global->LDS copy (ASYNCcnt path), with safe fallback.
// ---------------------------------------------------------------------------
#if __has_builtin(__builtin_amdgcn_global_load_async_to_lds_b128)
#define HAVE_ASYNC_LDS 1
#else
#define HAVE_ASYNC_LDS 0
#endif

__device__ __forceinline__ void async_cp16(const bf16* g, bf16* l) {
#if HAVE_ASYNC_LDS
    __builtin_amdgcn_global_load_async_to_lds_b128(
        (AS_GLOBAL i32x4*)g, (AS_LDS i32x4*)l, 0, 0);
#else
    *(bf16x8*)l = *(const bf16x8*)g;
#endif
}

__device__ __forceinline__ void wait_async0() {
#if HAVE_ASYNC_LDS
#if __has_builtin(__builtin_amdgcn_s_wait_asynccnt)
    __builtin_amdgcn_s_wait_asynccnt(0);
#else
    asm volatile("s_wait_asynccnt 0x0" ::: "memory");
#endif
#endif
}

// ---------------------------------------------------------------------------
// WMMA fragment loaders (CDNA5 wave32 layouts, ISA 7.12.2)
// ---------------------------------------------------------------------------

// A-matrix 16x32 bf16, row-major source (ld elements between rows).
// lane<16:  row=lane,    K = [0..7] and [16..23]
// lane>=16: row=lane-16, K = [8..15] and [24..31]
__device__ __forceinline__ bf16x16 load_frag_a(const bf16* base, int ld, int lane) {
    const bf16* p = base + (lane & 15) * ld + ((lane >> 4) << 3);
    bf16x8 lo = *(const bf16x8*)(p);
    bf16x8 hi = *(const bf16x8*)(p + 16);
    return __builtin_shufflevector(lo, hi, 0,1,2,3,4,5,6,7,8,9,10,11,12,13,14,15);
}

// B-matrix 32x16 bf16, source is B^T row-major (i.e. column-major B, ld elems).
// lane holds column n=lane&15, K = [8h..8h+7] and [16+8h..16+8h+7], h=lane>>4.
__device__ __forceinline__ bf16x16 load_frag_b(const bf16* base, int ld, int lane) {
    const bf16* p = base + (lane & 15) * ld + ((lane >> 4) << 3);
    bf16x8 lo = *(const bf16x8*)(p);
    bf16x8 hi = *(const bf16x8*)(p + 16);
    return __builtin_shufflevector(lo, hi, 0,1,2,3,4,5,6,7,8,9,10,11,12,13,14,15);
}

__device__ __forceinline__ f32x8 wmma_bf16(bf16x16 a, bf16x16 b, f32x8 c) {
    return __builtin_amdgcn_wmma_f32_16x16x32_bf16(false, a, false, b,
                                                   (short)0, c, false, false);
}

// ---------------------------------------------------------------------------
// Kernel 0: f32 -> bf16 conversion (grid-stride)
// ---------------------------------------------------------------------------
__global__ void cvt_f32_bf16(const float* __restrict__ in, bf16* __restrict__ out, long n) {
    long i = (long)blockIdx.x * blockDim.x + threadIdx.x;
    long stride = (long)gridDim.x * blockDim.x;
    for (; i < n; i += stride) out[i] = (bf16)in[i];
}

// Zero the K-pad columns [48,64) of v_ws (B,H,N,64)
__global__ void zero_vpad(bf16* __restrict__ v, int rows) {
    int r = blockIdx.x * blockDim.x + threadIdx.x;
    int stride = gridDim.x * blockDim.x;
    bf16x8 z = {};
    for (; r < rows; r += stride) {
        *(bf16x8*)(v + (long)r * 64 + 48) = z;
        *(bf16x8*)(v + (long)r * 64 + 56) = z;
    }
}

// ---------------------------------------------------------------------------
// Kernel 1: QKV GEMM.  (100352 x 384) @ (384 x 1152) -> scatter to q/k/v ws.
// Block tile 128x128, K-step 32, 8 waves each computing 32x64.
// Double-buffered LDS tiles filled by async global->LDS DMA.
// ---------------------------------------------------------------------------
#define BM 128
#define BN 128
#define BK 32

__global__ __launch_bounds__(256) void qkv_gemm(const bf16* __restrict__ X,
                                                const bf16* __restrict__ W,
                                                bf16* __restrict__ Q,
                                                bf16* __restrict__ Kw,
                                                bf16* __restrict__ V) {
    __shared__ bf16 sA[2][BM * BK];
    __shared__ bf16 sB[2][BN * BK];
    const int tid = threadIdx.x;
    const int lane = tid & 31, wave = tid >> 5;
    const int wm = wave >> 1, wn = wave & 1;           // 4x2 wave grid
    const long row0 = (long)blockIdx.x * BM;
    const int col0 = blockIdx.y * BN;

    f32x8 acc[2][4];
    #pragma unroll
    for (int i = 0; i < 2; i++)
        #pragma unroll
        for (int j = 0; j < 4; j++) acc[i][j] = (f32x8){};

    // Per-thread tile-load: 2 A-chunks + 2 B-chunks of 16B each (4 async ops).
    auto load_tile = [&](int kt, int buf) {
        #pragma unroll
        for (int s = tid; s < (BM * BK / 8); s += 256) {
            int r = s >> 2, c8 = (s & 3) << 3;
            async_cp16(&X[(row0 + r) * 384 + kt + c8], &sA[buf][r * BK + c8]);
        }
        #pragma unroll
        for (int s = tid; s < (BN * BK / 8); s += 256) {
            int r = s >> 2, c8 = (s & 3) << 3;
            async_cp16(&W[(long)(col0 + r) * 384 + kt + c8], &sB[buf][r * BK + c8]);
        }
    };

    load_tile(0, 0);
    int cur = 0;
    for (int kt = 0; kt < 384; kt += BK) {
        wait_async0();        // this wave's DMA into buf `cur` has landed
        __syncthreads();      // all waves done; prior reads of buf cur^1 drained
        if (kt + BK < 384) load_tile(kt + BK, cur ^ 1);

        bf16x16 af[2], bfr[4];
        #pragma unroll
        for (int i = 0; i < 2; i++)
            af[i] = load_frag_a(&sA[cur][(wm * 32 + i * 16) * BK], BK, lane);
        #pragma unroll
        for (int j = 0; j < 4; j++)
            bfr[j] = load_frag_b(&sB[cur][(wn * 64 + j * 16) * BK], BK, lane);
        #pragma unroll
        for (int i = 0; i < 2; i++)
            #pragma unroll
            for (int j = 0; j < 4; j++)
                acc[i][j] = wmma_bf16(af[i], bfr[j], acc[i][j]);
        cur ^= 1;
    }

    // Epilogue: decode output column into (qkv, h, c) and scatter.
    const int rlo = lane & 15, rhi = lane >> 4;
    #pragma unroll
    for (int i = 0; i < 2; i++)
        #pragma unroll
        for (int j = 0; j < 4; j++)
            #pragma unroll
            for (int r = 0; r < 8; r++) {
                long row = row0 + wm * 32 + i * 16 + r + (rhi << 3); // global token row
                int col = col0 + wn * 64 + j * 16 + rlo;             // 0..1151
                bf16 val = (bf16)acc[i][j][r];
                int which = col / 384, rem = col % 384;
                int h = rem / 48, c = rem % 48;
                long b = row / 3136, n = row % 3136;
                long bh = b * 8 + h;
                if (which == 0)      Q [(bh * 48 + c) * 3136 + n] = val;
                else if (which == 1) Kw[(bh * 48 + c) * 3136 + n] = val;
                else                 V [(bh * 3136 + n) * 64 + c] = val;
            }
}

// ---------------------------------------------------------------------------
// Kernel 2: fused XCA per (b,h): Gram(48x48) via WMMA + norms via VALU,
// softmax in LDS, out = attn @ v via WMMA -> (B,N,D) bf16.
// ---------------------------------------------------------------------------
__global__ __launch_bounds__(256) void xca_attn(const bf16* __restrict__ Qw,
                                                const bf16* __restrict__ Kw,
                                                const bf16* __restrict__ Vw,
                                                const float* __restrict__ temp,
                                                bf16* __restrict__ Mid) {
    __shared__ float sG[48 * 48];
    __shared__ float sSQ[96];
    __shared__ bf16 sAttn[48 * 64];

    const int bh = blockIdx.x;
    const int b = bh >> 3, h = bh & 7;
    const int tid = threadIdx.x, lane = tid & 31, wave = tid >> 5;
    const int rlo = lane & 15, rhi = lane >> 4;

    const bf16* q = Qw + (long)bh * 48 * 3136;
    const bf16* k = Kw + (long)bh * 48 * 3136;
    const bf16* v = Vw + (long)bh * 3136 * 64;

    for (int i = tid; i < 48 * 48; i += 256) sG[i] = 0.f;
    if (tid < 96) sSQ[tid] = 0.f;
    __syncthreads();

    // ---- Phase A: Gram matrix + sum-of-squares, streaming over N=3136 ----
    f32x8 g[3][3];
    #pragma unroll
    for (int a = 0; a < 3; a++)
        #pragma unroll
        for (int d = 0; d < 3; d++) g[a][d] = (f32x8){};
    float sqq[3] = {0.f, 0.f, 0.f}, sqk[3] = {0.f, 0.f, 0.f};

    for (int ch = wave; ch < 98; ch += 8) {       // 98 K-chunks of 32 tokens
        int n0 = ch << 5;
        if (ch + 8 < 98) {                         // prefetch next chunk
            int np = (ch + 8) << 5;
            #pragma unroll
            for (int t = 0; t < 3; t++) {
                __builtin_prefetch(q + (t * 16 + rlo) * 3136 + np + (rhi << 3), 0, 0);
                __builtin_prefetch(k + (t * 16 + rlo) * 3136 + np + (rhi << 4), 0, 0);
            }
        }
        bf16x16 aq[3], bk[3];
        #pragma unroll
        for (int t = 0; t < 3; t++) aq[t] = load_frag_a(q + (t * 16) * 3136 + n0, 3136, lane);
        #pragma unroll
        for (int t = 0; t < 3; t++) bk[t] = load_frag_b(k + (t * 16) * 3136 + n0, 3136, lane);
        #pragma unroll
        for (int ct = 0; ct < 3; ct++)
            #pragma unroll
            for (int dt = 0; dt < 3; dt++)
                g[ct][dt] = wmma_bf16(aq[ct], bk[dt], g[ct][dt]);
        #pragma unroll
        for (int t = 0; t < 3; t++)
            #pragma unroll
            for (int e = 0; e < 16; e++) {
                float fq = (float)aq[t][e]; sqq[t] += fq * fq;
                float fk = (float)bk[t][e]; sqk[t] += fk * fk;
            }
    }
    // Cross-wave reduction into LDS.
    #pragma unroll
    for (int ct = 0; ct < 3; ct++)
        #pragma unroll
        for (int dt = 0; dt < 3; dt++)
            #pragma unroll
            for (int r = 0; r < 8; r++)
                atomicAdd(&sG[(ct * 16 + r + (rhi << 3)) * 48 + dt * 16 + rlo], g[ct][dt][r]);
    #pragma unroll
    for (int t = 0; t < 3; t++) {
        atomicAdd(&sSQ[t * 16 + rlo], sqq[t]);
        atomicAdd(&sSQ[48 + t * 16 + rlo], sqk[t]);
    }
    __syncthreads();

    // ---- Phase B: scale by norms + temperature, then row softmax ----
    const float T = temp[h];
    for (int i = tid; i < 48 * 48; i += 256) {
        int c = i / 48, d = i % 48;
        float dq = fmaxf(sqrtf(sSQ[c]), 1e-12f);
        float dk = fmaxf(sqrtf(sSQ[48 + d]), 1e-12f);
        sG[i] = sG[i] * T / (dq * dk);
    }
    __syncthreads();
    if (tid < 48) {
        int c = tid;
        float mx = -3.4e38f;
        for (int d = 0; d < 48; d++) mx = fmaxf(mx, sG[c * 48 + d]);
        float sum = 0.f;
        for (int d = 0; d < 48; d++) sum += __expf(sG[c * 48 + d] - mx);
        float inv = 1.f / sum;
        for (int d = 0; d < 48; d++)
            sAttn[c * 64 + d] = (bf16)(__expf(sG[c * 48 + d] - mx) * inv);
        #pragma unroll
        for (int d = 48; d < 64; d++) sAttn[c * 64 + d] = (bf16)0.f;  // zero K-pad
    }
    __syncthreads();

    // ---- Phase C: out^T = v^T (3136x64) @ attn^T (64x48), K padded to 64 ----
    bf16x16 bfr[2][3];
    #pragma unroll
    for (int ks = 0; ks < 2; ks++)
        #pragma unroll
        for (int nt = 0; nt < 3; nt++)
            bfr[ks][nt] = load_frag_b(&sAttn[(nt * 16) * 64 + ks * 32], 64, lane);

    for (int mt = wave; mt < 196; mt += 8) {      // 196 token tiles of 16
        if (mt + 8 < 196)
            __builtin_prefetch(v + ((mt + 8) * 16 + rlo) * 64 + (rhi << 3), 0, 0);
        bf16x16 av[2];
        #pragma unroll
        for (int ks = 0; ks < 2; ks++)
            av[ks] = load_frag_a(v + (mt * 16) * 64 + ks * 32, 64, lane);
        f32x8 o[3] = {(f32x8){}, (f32x8){}, (f32x8){}};
        #pragma unroll
        for (int ks = 0; ks < 2; ks++)
            #pragma unroll
            for (int nt = 0; nt < 3; nt++)
                o[nt] = wmma_bf16(av[ks], bfr[ks][nt], o[nt]);
        long tokBase = (long)b * 3136 + mt * 16;
        #pragma unroll
        for (int nt = 0; nt < 3; nt++)
            #pragma unroll
            for (int r = 0; r < 8; r++) {
                long tok = tokBase + r + (rhi << 3);
                int col = h * 48 + nt * 16 + rlo;
                Mid[tok * 384 + col] = (bf16)o[nt][r];
            }
    }
}

// ---------------------------------------------------------------------------
// Kernel 3: output projection. (100352 x 384) @ (384 x 384) + bias -> f32.
// Same double-buffered async-LDS mainloop as the QKV GEMM.
// ---------------------------------------------------------------------------
__global__ __launch_bounds__(256) void proj_gemm(const bf16* __restrict__ A,
                                                 const bf16* __restrict__ W,
                                                 const float* __restrict__ bias,
                                                 float* __restrict__ Out) {
    __shared__ bf16 sA[2][BM * BK];
    __shared__ bf16 sB[2][BN * BK];
    const int tid = threadIdx.x;
    const int lane = tid & 31, wave = tid >> 5;
    const int wm = wave >> 1, wn = wave & 1;
    const long row0 = (long)blockIdx.x * BM;
    const int col0 = blockIdx.y * BN;

    f32x8 acc[2][4];
    #pragma unroll
    for (int i = 0; i < 2; i++)
        #pragma unroll
        for (int j = 0; j < 4; j++) acc[i][j] = (f32x8){};

    auto load_tile = [&](int kt, int buf) {
        #pragma unroll
        for (int s = tid; s < (BM * BK / 8); s += 256) {
            int r = s >> 2, c8 = (s & 3) << 3;
            async_cp16(&A[(row0 + r) * 384 + kt + c8], &sA[buf][r * BK + c8]);
        }
        #pragma unroll
        for (int s = tid; s < (BN * BK / 8); s += 256) {
            int r = s >> 2, c8 = (s & 3) << 3;
            async_cp16(&W[(long)(col0 + r) * 384 + kt + c8], &sB[buf][r * BK + c8]);
        }
    };

    load_tile(0, 0);
    int cur = 0;
    for (int kt = 0; kt < 384; kt += BK) {
        wait_async0();
        __syncthreads();
        if (kt + BK < 384) load_tile(kt + BK, cur ^ 1);

        bf16x16 af[2], bfr[4];
        #pragma unroll
        for (int i = 0; i < 2; i++)
            af[i] = load_frag_a(&sA[cur][(wm * 32 + i * 16) * BK], BK, lane);
        #pragma unroll
        for (int j = 0; j < 4; j++)
            bfr[j] = load_frag_b(&sB[cur][(wn * 64 + j * 16) * BK], BK, lane);
        #pragma unroll
        for (int i = 0; i < 2; i++)
            #pragma unroll
            for (int j = 0; j < 4; j++)
                acc[i][j] = wmma_bf16(af[i], bfr[j], acc[i][j]);
        cur ^= 1;
    }

    const int rlo = lane & 15, rhi = lane >> 4;
    #pragma unroll
    for (int i = 0; i < 2; i++)
        #pragma unroll
        for (int j = 0; j < 4; j++)
            #pragma unroll
            for (int r = 0; r < 8; r++) {
                long row = row0 + wm * 32 + i * 16 + r + (rhi << 3);
                int col = col0 + wn * 64 + j * 16 + rlo;
                Out[row * 384 + col] = acc[i][j][r] + bias[col];
            }
}

// ---------------------------------------------------------------------------
// Launch
// ---------------------------------------------------------------------------
extern "C" void kernel_launch(void* const* d_in, const int* in_sizes, int n_in,
                              void* d_out, int out_size, void* d_ws, size_t ws_size,
                              hipStream_t stream) {
    const float* x      = (const float*)d_in[0];   // (32,3136,384)
    const float* qkv_w  = (const float*)d_in[1];   // (1152,384)
    const float* temp   = (const float*)d_in[2];   // (8,1,1)
    const float* proj_w = (const float*)d_in[3];   // (384,384)
    const float* proj_b = (const float*)d_in[4];   // (384,)
    float* out = (float*)d_out;

    char* ws = (char*)d_ws;
    bf16* X16   = (bf16*)(ws);                 //  77,070,336 B
    bf16* WQ16  = (bf16*)(ws + 77070336L);     //     884,736 B
    bf16* WP16  = (bf16*)(ws + 77955072L);     //     294,912 B
    bf16* Q16   = (bf16*)(ws + 78249984L);     //  77,070,336 B   (B,H,C,N)
    bf16* K16   = (bf16*)(ws + 155320320L);    //  77,070,336 B   (B,H,C,N)
    bf16* V16   = (bf16*)(ws + 232390656L);    // 102,760,448 B   (B,H,N,64)
    bf16* MID16 = (bf16*)(ws + 335151104L);    //  77,070,336 B   (B,N,D)

    cvt_f32_bf16<<<2048, 256, 0, stream>>>(x, X16, 38535168L);
    cvt_f32_bf16<<<64,   256, 0, stream>>>(qkv_w, WQ16, 442368L);
    cvt_f32_bf16<<<32,   256, 0, stream>>>(proj_w, WP16, 147456L);
    zero_vpad   <<<1024, 256, 0, stream>>>(V16, 802816);

    qkv_gemm <<<dim3(784, 9), 256, 0, stream>>>(X16, WQ16, Q16, K16, V16);
    xca_attn <<<dim3(256),    256, 0, stream>>>(Q16, K16, V16, temp, MID16);
    proj_gemm<<<dim3(784, 3), 256, 0, stream>>>(MID16, WP16, proj_b, out);
}
